// AttentionDecoder_2662879723658
// MI455X (gfx1250) — matrix operand order repaired and checked by reading, not compiled, not run on hardware
//
#include <hip/hip_runtime.h>
#include <math.h>

typedef __bf16 bf16;
typedef bf16  bf16x8  __attribute__((ext_vector_type(8)));
typedef bf16  bf16x16 __attribute__((ext_vector_type(16)));
typedef float f32x8   __attribute__((ext_vector_type(8)));

__device__ __forceinline__ bf16x16 load_afrag(const bf16* ap)
{
  bf16x8 lo = *(const bf16x8*)(ap);
  bf16x8 hi = *(const bf16x8*)(ap + 8);
  bf16x16 v;
#pragma unroll
  for (int j = 0; j < 8; ++j) { v[j] = lo[j]; v[8 + j] = hi[j]; }
  return v;
}

// One K-step worth of fragments: 1 B fragment + 4 A fragments (64 rows).
struct Frags {
  bf16x16 b;
  bf16x16 a[4];
};

__device__ __forceinline__ Frags load_frags(const bf16* brow, const bf16* a0,
                                            size_t astep, int k0)
{
  Frags f;
  f.b = *(const bf16x16*)(brow + k0);
#pragma unroll
  for (int i = 0; i < 4; ++i) f.a[i] = load_afrag(a0 + k0 + astep * i);
  return f;
}

__device__ __forceinline__ void compute_frags(f32x8 acc[4], const Frags& f)
{
#pragma unroll
  for (int i = 0; i < 4; ++i)
    acc[i] = __builtin_amdgcn_wmma_f32_16x16x32_bf16(
        false, f.a[i], false, f.b, (short)0, acc[i], false, false);
}

// ---------------------------------------------------------------------------
// WMMA GEMM:  C(MxN) = A(MxK) @ B(KxN) + bias, A bf16 row-major (lda),
// Bt = B^T bf16 (N rows x K cols, rows 32B aligned), C fp32 (ldc),
// optional bf16 mirror of C, optional sigmoid.
// Each wave computes a 64-row x 16-col strip: 4 f32x8 accumulators,
// one B fragment reused by 4 v_wmma_f32_16x16x32_bf16 per K-step.
// K-loop is software-pipelined with unroll-by-2 modulo variable expansion:
// two fragment sets alternate compute/refill so loads overlap the matrix
// pipe with NO register rotation copies.
// M % 64 == 0, N % 16 == 0, K % 64 == 0, K >= 64. Whole waves retire
// together -> EXEC is all ones at every WMMA.
// ---------------------------------------------------------------------------
__global__ __launch_bounds__(128) void wmma_gemm_bf16(
    const bf16* __restrict__ A, int lda,
    const bf16* __restrict__ Bt,
    const float* __restrict__ bias,
    float* __restrict__ C, int ldc,
    bf16* __restrict__ Cbf, int ldcbf,
    int M, int N, int K, int act)
{
  const int lane = threadIdx.x & 31;
  const int wave = threadIdx.x >> 5;
  const int nTiles = N >> 4;
  const int strips = (M >> 6) * nTiles;
  const int s = blockIdx.x * 4 + wave;
  if (s >= strips) return;
  const int m0 = (s / nTiles) << 6;
  const int n0 = (s % nTiles) << 4;

  const int half = lane >> 4;
  const int l15  = lane & 15;

  f32x8 acc[4];
#pragma unroll
  for (int i = 0; i < 4; ++i) acc[i] = (f32x8)0.f;

  // B fragment source: row (n0+l15) of Bt, K offset 16*half (contiguous 16 bf16)
  const bf16* brow = Bt + (size_t)(n0 + l15) * K + 16 * half;
  // A fragment source: row (m0 + i*16 + l15), K offset 16*half (contiguous 16 bf16)
  const bf16* a0 = A + (size_t)(m0 + l15) * lda + 16 * half;
  const size_t astep = (size_t)16 * lda;

  // ---- prologue: fill both pipeline stages ----
  Frags fA = load_frags(brow, a0, astep, 0);
  Frags fB = load_frags(brow, a0, astep, 32);

  // ---- unrolled-by-2 pipelined loop: consume a stage, refill it in place ----
  int k = 64;
  for (; k + 64 <= K; k += 64) {
    compute_frags(acc, fA);
    __builtin_prefetch(brow + k + 64, 0, 1);
    fA = load_frags(brow, a0, astep, k);
    compute_frags(acc, fB);
    fB = load_frags(brow, a0, astep, k + 32);
  }

  // ---- epilogue: drain both stages ----
  compute_frags(acc, fA);
  compute_frags(acc, fB);

  const int col = n0 + l15;
  const float bcol = bias ? bias[col] : 0.f;
#pragma unroll
  for (int i = 0; i < 4; ++i) {
#pragma unroll
    for (int j = 0; j < 8; ++j) {
      int row = m0 + i * 16 + j + 8 * half;  // C layout: m = j + 8*half
      float v = acc[i][j] + bcol;
      if (act == 1) v = 1.f / (1.f + expf(-v));
      C[(size_t)row * ldc + col] = v;
      if (Cbf) Cbf[(size_t)row * ldcbf + col] = (bf16)v;
    }
  }
}

// ---------------------------------------------------------------------------
// One-time conversion helpers
// ---------------------------------------------------------------------------
__global__ void convert_f32_bf16(const float* __restrict__ src,
                                 bf16* __restrict__ dst, int n)
{
  int i = blockIdx.x * blockDim.x + threadIdx.x;
  if (i < n) dst[i] = (bf16)src[i];
}

// src is (K,N) row-major fp32; dst[n*dstStride + dstOff + k] = bf16(src[k,n])
__global__ void transpose_f32_bf16(const float* __restrict__ src,
                                   bf16* __restrict__ dst,
                                   int K, int N, int dstStride, int dstOff)
{
  int i = blockIdx.x * blockDim.x + threadIdx.x;
  if (i >= K * N) return;
  int k = i / N, n = i % N;
  dst[(size_t)n * dstStride + dstOff + k] = (bf16)src[i];
}

__global__ void bias_sum(const float* __restrict__ a,
                         const float* __restrict__ b,
                         float* __restrict__ o, int n)
{
  int i = blockIdx.x * blockDim.x + threadIdx.x;
  if (i < n) o[i] = a[i] + b[i];
}

// mean over P of encoder_out -> bf16 (B x 512), one block per batch row
__global__ __launch_bounds__(256) void mean_feat_kernel(
    const float* __restrict__ enc, bf16* __restrict__ mean_bf)
{
  int b = blockIdx.x, tid = threadIdx.x;
  for (int e = tid; e < 512; e += 256) {
    float acc = 0.f;
    const float* eb = enc + ((size_t)b * 196) * 512 + e;
    for (int p = 0; p < 196; ++p) acc += eb[(size_t)p * 512];
    mean_bf[(size_t)b * 512 + e] = (bf16)(acc * (1.f / 196.f));
  }
}

// ---------------------------------------------------------------------------
// Attention: energy -> softmax -> alpha (to d_out) + context.
// One block (256 threads) per batch row.
// ---------------------------------------------------------------------------
__global__ __launch_bounds__(256) void attention_step(
    const float* __restrict__ att_enc,  // B x 196 x 512
    const float* __restrict__ att_dec,  // B x 512
    const float* __restrict__ enc,      // B x 196 x 512
    const float* __restrict__ Wfull,    // 512
    const float* __restrict__ bfull,    // 1
    float* __restrict__ alphas,         // B x T x 196
    float* __restrict__ ctx,            // B x 512
    int t, int T)
{
  __shared__ float sdec[512];
  __shared__ float swf[512];
  __shared__ float sprob[196];
  __shared__ float sred[256];
  const int b = blockIdx.x, tid = threadIdx.x;

  for (int a = tid; a < 512; a += 256) {
    sdec[a] = att_dec[b * 512 + a];
    swf[a]  = Wfull[a];
  }
  __syncthreads();

  for (int p = tid; p < 196; p += 256) {
    const float* ae = att_enc + ((size_t)b * 196 + p) * 512;
    float e = 0.f;
#pragma unroll 4
    for (int a = 0; a < 512; ++a) e += fmaxf(ae[a] + sdec[a], 0.f) * swf[a];
    sprob[p] = e + bfull[0];
  }
  __syncthreads();

  float lm = -3.4e38f;
  for (int p = tid; p < 196; p += 256) lm = fmaxf(lm, sprob[p]);
  sred[tid] = lm; __syncthreads();
  for (int s = 128; s > 0; s >>= 1) {
    if (tid < s) sred[tid] = fmaxf(sred[tid], sred[tid + s]);
    __syncthreads();
  }
  const float mx = sred[0];
  __syncthreads();

  float ls = 0.f;
  for (int p = tid; p < 196; p += 256) {
    float e = expf(sprob[p] - mx);
    sprob[p] = e;
    ls += e;
  }
  sred[tid] = ls; __syncthreads();
  for (int s = 128; s > 0; s >>= 1) {
    if (tid < s) sred[tid] += sred[tid + s];
    __syncthreads();
  }
  const float inv = 1.f / sred[0];

  for (int p = tid; p < 196; p += 256) {
    float a = sprob[p] * inv;
    sprob[p] = a;
    alphas[((size_t)b * T + t) * 196 + p] = a;
  }
  __syncthreads();

  for (int e = tid; e < 512; e += 256) {
    float acc = 0.f;
    const float* eb = enc + ((size_t)b * 196) * 512 + e;
    for (int p = 0; p < 196; ++p) acc += sprob[p] * eb[(size_t)p * 512];
    ctx[(size_t)b * 512 + e] = acc;
  }
}

// x = [emb_t (1024) | gate*context (512) | h (1024)] -> bf16 (64 x 2560)
__global__ void build_xfull(const float* __restrict__ emb,
                            const int* __restrict__ cap,
                            const float* __restrict__ gate,
                            const float* __restrict__ ctx,
                            const bf16* __restrict__ h_bf,
                            bf16* __restrict__ xfull, int t, int L)
{
  int idx = blockIdx.x * blockDim.x + threadIdx.x;
  if (idx >= 64 * 2560) return;
  int b = idx / 2560, col = idx % 2560;
  bf16 v;
  if (col < 1024) {
    int tok = cap[b * L + t];
    v = (bf16)emb[(size_t)tok * 1024 + col];
  } else if (col < 1536) {
    int a = col - 1024;
    v = (bf16)(gate[b * 512 + a] * ctx[b * 512 + a]);
  } else {
    v = h_bf[b * 1024 + (col - 1536)];
  }
  xfull[idx] = v;
}

// LSTM pointwise cell: gates (64x4096, [i f g o]) + c -> c, h(bf16)
__global__ void lstm_cell(const float* __restrict__ gates,
                          float* __restrict__ c,
                          bf16* __restrict__ h_bf)
{
  int idx = blockIdx.x * blockDim.x + threadIdx.x;
  if (idx >= 64 * 1024) return;
  int b = idx >> 10, j = idx & 1023;
  const float* g = gates + (size_t)b * 4096;
  float ig = 1.f / (1.f + expf(-g[j]));
  float fg = 1.f / (1.f + expf(-g[1024 + j]));
  float gg = tanhf(g[2048 + j]);
  float og = 1.f / (1.f + expf(-g[3072 + j]));
  float cn = fg * c[idx] + ig * gg;
  c[idx] = cn;
  h_bf[idx] = (bf16)(og * tanhf(cn));
}

// ---------------------------------------------------------------------------
extern "C" void kernel_launch(void* const* d_in, const int* in_sizes, int n_in,
                              void* d_out, int out_size, void* d_ws, size_t ws_size,
                              hipStream_t stream)
{
  const int B = 64, P = 196, ENC = 512, H = 1024, Adim = 512, V = 20000;
  const int L = in_sizes[1] / B;   // captions is (B, L)
  const int T = L - 1;
  const int KCAT = 2560;           // 1024 emb + 512 gated ctx + 1024 h

  const float* enc       = (const float*)d_in[0];
  const int*   captions  = (const int*)d_in[1];
  const float* W_enc_att = (const float*)d_in[3];
  const float* b_enc_att = (const float*)d_in[4];
  const float* W_dec_att = (const float*)d_in[5];
  const float* b_dec_att = (const float*)d_in[6];
  const float* W_full    = (const float*)d_in[7];
  const float* b_full    = (const float*)d_in[8];
  const float* emb       = (const float*)d_in[9];
  const float* W_ih      = (const float*)d_in[10];
  const float* b_ih      = (const float*)d_in[11];
  const float* W_hh      = (const float*)d_in[12];
  const float* b_hh      = (const float*)d_in[13];
  const float* W_init_h  = (const float*)d_in[14];
  const float* b_init_h  = (const float*)d_in[15];
  const float* W_init_c  = (const float*)d_in[16];
  const float* b_init_c  = (const float*)d_in[17];
  const float* W_fbeta   = (const float*)d_in[18];
  const float* b_fbeta   = (const float*)d_in[19];
  const float* W_fc      = (const float*)d_in[20];
  const float* b_fc      = (const float*)d_in[21];

  char* wptr = (char*)d_ws;
  auto alloc = [&](size_t bytes) -> void* {
    void* p = wptr;
    wptr += (bytes + 255) & ~(size_t)255;
    return p;
  };

  bf16*  enc_bf   = (bf16*)alloc((size_t)B * P * ENC * 2);
  bf16*  Wenc_t   = (bf16*)alloc((size_t)Adim * ENC * 2);   // 512 x 512 (NxK)
  bf16*  Wdec_t   = (bf16*)alloc((size_t)Adim * H * 2);     // 512 x 1024
  bf16*  Wfb_t    = (bf16*)alloc((size_t)ENC * H * 2);      // 512 x 1024
  bf16*  Winh_t   = (bf16*)alloc((size_t)H * ENC * 2);      // 1024 x 512
  bf16*  Winc_t   = (bf16*)alloc((size_t)H * ENC * 2);
  bf16*  Wcat_t   = (bf16*)alloc((size_t)4 * H * KCAT * 2); // 4096 x 2560
  bf16*  Wfc_t    = (bf16*)alloc((size_t)V * H * 2);        // 20000 x 1024
  float* bias_cat = (float*)alloc((size_t)4 * H * 4);
  float* att_enc  = (float*)alloc((size_t)B * P * Adim * 4);
  bf16*  mean_bf  = (bf16*)alloc((size_t)B * ENC * 2);
  float* hbuf     = (float*)alloc((size_t)B * H * 4);
  float* cbuf     = (float*)alloc((size_t)B * H * 4);
  bf16*  h_bf     = (bf16*)alloc((size_t)B * H * 2);
  float* attdec   = (float*)alloc((size_t)B * Adim * 4);
  float* gatebuf  = (float*)alloc((size_t)B * ENC * 4);
  float* ctx      = (float*)alloc((size_t)B * ENC * 4);
  bf16*  xfull    = (bf16*)alloc((size_t)B * KCAT * 2);
  float* gates    = (float*)alloc((size_t)B * 4 * H * 4);

  float* preds  = (float*)d_out;                     // (B, T, V)
  float* alphas = preds + (size_t)B * T * V;         // (B, T, P)

  auto cdiv = [](long a, long b) { return (int)((a + b - 1) / b); };

  auto gemm = [&](const bf16* Abf, int lda, const bf16* Bt, const float* bias,
                  float* C, int ldc, bf16* Cbf, int ldcbf,
                  int M, int N, int K, int act) {
    int strips = (M / 64) * (N / 16);
    int blocks = (strips + 3) / 4;  // 4 waves (128 threads) per block
    wmma_gemm_bf16<<<blocks, 128, 0, stream>>>(Abf, lda, Bt, bias, C, ldc,
                                               Cbf, ldcbf, M, N, K, act);
  };

  // ---- one-time conversions (recomputed every call for determinism) ----
  convert_f32_bf16<<<cdiv((long)B * P * ENC, 256), 256, 0, stream>>>(
      enc, enc_bf, B * P * ENC);
  transpose_f32_bf16<<<cdiv((long)ENC * Adim, 256), 256, 0, stream>>>(
      W_enc_att, Wenc_t, ENC, Adim, ENC, 0);
  transpose_f32_bf16<<<cdiv((long)H * Adim, 256), 256, 0, stream>>>(
      W_dec_att, Wdec_t, H, Adim, H, 0);
  transpose_f32_bf16<<<cdiv((long)H * ENC, 256), 256, 0, stream>>>(
      W_fbeta, Wfb_t, H, ENC, H, 0);
  transpose_f32_bf16<<<cdiv((long)ENC * H, 256), 256, 0, stream>>>(
      W_init_h, Winh_t, ENC, H, ENC, 0);
  transpose_f32_bf16<<<cdiv((long)ENC * H, 256), 256, 0, stream>>>(
      W_init_c, Winc_t, ENC, H, ENC, 0);
  transpose_f32_bf16<<<cdiv((long)1536 * 4 * H, 256), 256, 0, stream>>>(
      W_ih, Wcat_t, 1536, 4 * H, KCAT, 0);
  transpose_f32_bf16<<<cdiv((long)H * 4 * H, 256), 256, 0, stream>>>(
      W_hh, Wcat_t, H, 4 * H, KCAT, 1536);
  transpose_f32_bf16<<<cdiv((long)H * V, 256), 256, 0, stream>>>(
      W_fc, Wfc_t, H, V, H, 0);
  bias_sum<<<cdiv(4 * H, 256), 256, 0, stream>>>(b_ih, b_hh, bias_cat, 4 * H);
  mean_feat_kernel<<<B, 256, 0, stream>>>(enc, mean_bf);

  // att_enc = encoder_out @ W_enc_att + b  (12544 x 512, K=512)
  gemm(enc_bf, ENC, Wenc_t, b_enc_att, att_enc, Adim, nullptr, 0,
       B * P, Adim, ENC, 0);
  // h0 (also mirrored to bf16), c0
  gemm(mean_bf, ENC, Winh_t, b_init_h, hbuf, H, h_bf, H, B, H, ENC, 0);
  gemm(mean_bf, ENC, Winc_t, b_init_c, cbuf, H, nullptr, 0, B, H, ENC, 0);

  // ---- sequential decode ----
  for (int t = 0; t < T; ++t) {
    // att_dec = h @ W_dec_att + b
    gemm(h_bf, H, Wdec_t, b_dec_att, attdec, Adim, nullptr, 0, B, Adim, H, 0);
    // gate = sigmoid(h @ W_fbeta + b)
    gemm(h_bf, H, Wfb_t, b_fbeta, gatebuf, ENC, nullptr, 0, B, ENC, H, 1);
    // softmax attention + context, alpha straight into d_out
    attention_step<<<B, 256, 0, stream>>>(att_enc, attdec, enc, W_full,
                                          b_full, alphas, ctx, t, T);
    // x = [emb_t | gate*ctx | h]
    build_xfull<<<cdiv((long)B * KCAT, 256), 256, 0, stream>>>(
        emb, captions, gatebuf, ctx, h_bf, xfull, t, L);
    // gates = x @ [W_ih; W_hh] + (b_ih + b_hh)
    gemm(xfull, KCAT, Wcat_t, bias_cat, gates, 4 * H, nullptr, 0,
         B, 4 * H, KCAT, 0);
    // c, h update (h mirrored to bf16 for next-step GEMMs)
    lstm_cell<<<cdiv((long)B * H, 256), 256, 0, stream>>>(gates, cbuf, h_bf);
    // preds_t = h @ W_fc + b_fc, written directly into d_out (ldc = T*V)
    gemm(h_bf, H, Wfc_t, b_fc, preds + (size_t)t * V, T * V, nullptr, 0,
         B, V, H, 0);
  }
}
